// HierarchicalEncoder2_64244120814203
// MI455X (gfx1250) — matrix operand order repaired and checked by reading, not compiled, not run on hardware
//
#include <hip/hip_runtime.h>
#include <hip/hip_bf16.h>

#define DEV __device__ __forceinline__

// Flip to 0 if the integrated assembler rejects the async-to-LDS mnemonics.
#define USE_ASYNC_STAGE 1

// ---------------- types ----------------
typedef __attribute__((ext_vector_type(16))) __bf16        v16bf;
typedef __attribute__((ext_vector_type(8)))  float         v8f;
typedef __attribute__((ext_vector_type(4)))  unsigned int  u32x4;

union BF16x16 {
    u32x4          q[2];
    v16bf          v;
    unsigned short s[16];
};

// ---------------- problem constants ----------------
#define NB 4
#define NN 2000
#define NK 32
#define NH 128
#define NR (NB * NN)              // 8000 nodes
#define NE (NR * NK)              // 256000 edges
#define SCALE_INV (1.0f / 30.0f)
#define LN_EPS 1e-6f

// ---------------- helpers ----------------
// Native hardware f32->bf16 convert (RNE): lowers to v_cvt_pk_bf16_f32 pairs.
DEV unsigned short f2bf(float f) {
    union { __bf16 h; unsigned short u; } x;
    x.h = (__bf16)f;
    return x.u;
}

DEV v8f v8f_zero() { v8f z = {0.f,0.f,0.f,0.f,0.f,0.f,0.f,0.f}; return z; }

DEV u32x4 ld16(const unsigned short* p) { return *(const u32x4*)p; }

DEV v8f wmma_bf16(const BF16x16& a, const BF16x16& b, v8f c) {
    // D(16x16,f32) = A(16x32,bf16) x B(32x16,bf16) + C
    return __builtin_amdgcn_wmma_f32_16x16x32_bf16(
        /*neg_a=*/false, a.v, /*neg_b=*/false, b.v,
        /*c_mod=*/(short)0, c, /*reuse_a=*/false, /*reuse_b=*/false);
}

// Build A tile (16x32 bf16) from an f32 row: first 8 at col o, next 8 at col o+16
DEV void a_from_f32(BF16x16& a, const float* row, int o) {
#pragma unroll
    for (int i = 0; i < 8; ++i) a.v[i]     = (__bf16)row[o + i];
#pragma unroll
    for (int i = 0; i < 8; ++i) a.v[8 + i] = (__bf16)row[o + 16 + i];
}

// Async DMA: 8KB weight chunk (global -> LDS), 16B per thread per step.
// ISA: the instruction offset is added to both the LDS and the global address,
// so one base pair + offset immediates covers all 4 steps. Tracked by ASYNCcnt.
DEV void stage8k(unsigned short* lds_dst, const unsigned short* gsrc, int tid) {
#if USE_ASYNC_STAGE
    unsigned lds_addr = (unsigned)(size_t)lds_dst + (unsigned)(tid * 16);
    unsigned long long gaddr = (unsigned long long)(size_t)gsrc + (unsigned)(tid * 16);
    asm volatile("global_load_async_to_lds_b128 %0, %1, off"
                 :: "v"(lds_addr), "v"(gaddr) : "memory");
    asm volatile("global_load_async_to_lds_b128 %0, %1, off offset:2048"
                 :: "v"(lds_addr), "v"(gaddr) : "memory");
    asm volatile("global_load_async_to_lds_b128 %0, %1, off offset:4096"
                 :: "v"(lds_addr), "v"(gaddr) : "memory");
    asm volatile("global_load_async_to_lds_b128 %0, %1, off offset:6144"
                 :: "v"(lds_addr), "v"(gaddr) : "memory");
    asm volatile("s_wait_asynccnt 0" ::: "memory");
#else
    const u32x4* gs = (const u32x4*)gsrc;
    u32x4* sd = (u32x4*)lds_dst;
#pragma unroll
    for (int i = 0; i < 4; ++i) sd[tid + i * 128] = gs[tid + i * 128];
#endif
}

// ====================================================================
// Weight pre-pack: W[Kd][128] f32 (row-major, input-dim major) -> bf16
// B-tile layout: dst[((kc*8+nc)*32 + L)*16 + j] = W[kc*32 + (L>=16)*8 + (j&7) + (j>=8)*16][nc*16 + (L&15)]
// ====================================================================
__global__ void pack_w_kernel(const float* __restrict__ src,
                              unsigned short* __restrict__ dst, int kchunks) {
    int t = blockIdx.x * blockDim.x + threadIdx.x;
    int total = kchunks * 8 * 32 * 16;
    if (t >= total) return;
    int j  = t & 15;
    int L  = (t >> 4) & 31;
    int nc = (t >> 9) & 7;
    int kc = t >> 12;
    int col  = nc * 16 + (L & 15);
    int krow = kc * 32 + ((L >> 4) << 3) + (j & 7) + ((j >> 3) << 4);
    dst[t] = f2bf(src[(size_t)krow * NH + col]);
}

// f32 -> bf16 elementwise
__global__ void cvt_bf16_kernel(const float* __restrict__ src,
                                unsigned short* __restrict__ dst, int n) {
    int t = blockIdx.x * blockDim.x + threadIdx.x;
    if (t < n) dst[t] = f2bf(src[t]);
}

// vmask[node][k] = mask[b, idx]; sumv[node] = sum_k vmask
__global__ void mask_setup_kernel(const int* __restrict__ eidx,
                                  const float* __restrict__ mask,
                                  float* __restrict__ vmask,
                                  float* __restrict__ sumv) {
    int node = blockIdx.x;
    int lane = threadIdx.x;        // 32 threads
    int bIdx = node / NN;
    int iv   = eidx[node * NK + lane];
    float v  = mask[bIdx * NN + iv];
    vmask[node * NK + lane] = v;
    float s = v;
    s += __shfl_xor(s, 16); s += __shfl_xor(s, 8);
    s += __shfl_xor(s, 4);  s += __shfl_xor(s, 2); s += __shfl_xor(s, 1);
    if (lane == 0) sumv[node] = s;
}

// ====================================================================
// proj_ln: Y = LN(X @ W + b) with ddof=1.  X f32 [M][128], W packed bf16.
// Block = 4 waves, each wave owns 32 rows (2 M-tiles x 8 N-tiles).
// ====================================================================
__global__ __launch_bounds__(128) void proj_ln_kernel(
    const float* __restrict__ X, int Mrows,
    const unsigned short* __restrict__ Wp,
    const float* __restrict__ bias, const float* __restrict__ gamma,
    const float* __restrict__ beta,
    float* __restrict__ Yf, unsigned short* __restrict__ Yb) {
    const int wave = threadIdx.x >> 5, lane = threadIdx.x & 31;
    const int r0 = (blockIdx.x * 4 + wave) * 32;
    if (r0 >= Mrows) return;
    const int lm = lane & 15, hi = lane >> 4, o1 = hi * 8;

    v8f acc[2][8];
#pragma unroll
    for (int m = 0; m < 2; ++m)
#pragma unroll
        for (int nc = 0; nc < 8; ++nc) acc[m][nc] = v8f_zero();

    const float* xr0 = X + (size_t)(r0 + lm) * NH;
    const float* xr1 = X + (size_t)(r0 + 16 + lm) * NH;
#pragma unroll
    for (int kc = 0; kc < 4; ++kc) {
        const int co = kc * 32 + o1;
        BF16x16 a0, a1;
        a_from_f32(a0, xr0, co);
        a_from_f32(a1, xr1, co);
#pragma unroll
        for (int nc = 0; nc < 8; ++nc) {
            BF16x16 bt;
            const unsigned short* wp = Wp + (size_t)((kc * 8 + nc) * 32 + lane) * 16;
            bt.q[0] = ld16(wp); bt.q[1] = ld16(wp + 8);
            acc[0][nc] = wmma_bf16(a0, bt, acc[0][nc]);
            acc[1][nc] = wmma_bf16(a1, bt, acc[1][nc]);
        }
    }
    float bv[8], gv[8], bev[8];
#pragma unroll
    for (int nc = 0; nc < 8; ++nc) {
        int c = nc * 16 + lm;
        bv[nc] = bias[c]; gv[nc] = gamma[c]; bev[nc] = beta[c];
    }
#pragma unroll
    for (int m = 0; m < 2; ++m)
#pragma unroll
        for (int nc = 0; nc < 8; ++nc)
#pragma unroll
            for (int v = 0; v < 8; ++v) acc[m][nc][v] += bv[nc];
    // LayerNorm per row: a row's 128 columns live in one 16-lane half (8 N-tiles)
#pragma unroll
    for (int m = 0; m < 2; ++m) {
#pragma unroll
        for (int v = 0; v < 8; ++v) {
            float s = 0.f;
#pragma unroll
            for (int nc = 0; nc < 8; ++nc) s += acc[m][nc][v];
            s += __shfl_xor(s, 1); s += __shfl_xor(s, 2);
            s += __shfl_xor(s, 4); s += __shfl_xor(s, 8);
            const float mu = s * (1.0f / 128.0f);
            float q = 0.f;
#pragma unroll
            for (int nc = 0; nc < 8; ++nc) { float d = acc[m][nc][v] - mu; q += d * d; }
            q += __shfl_xor(q, 1); q += __shfl_xor(q, 2);
            q += __shfl_xor(q, 4); q += __shfl_xor(q, 8);
            const float sig  = sqrtf(q * (1.0f / 127.0f) + LN_EPS);
            const float rstd = 1.0f / (sig + LN_EPS);
            const int row = r0 + m * 16 + hi * 8 + v;
#pragma unroll
            for (int nc = 0; nc < 8; ++nc) {
                const int c = nc * 16 + lm;
                const float y = gv[nc] * (acc[m][nc][v] - mu) * rstd + bev[nc];
                if (Yf) Yf[(size_t)row * NH + c] = y;
                if (Yb) Yb[(size_t)row * NH + c] = f2bf(y);
            }
        }
    }
}

// ====================================================================
// msg: per node, h_ev[32,512] = [h | h[idx] | hS[idx] | h_e]  (bf16 gathers)
//   m1 = relu(h_ev @ W1 + b1)        (16 K-chunks, W1 async-staged in LDS)
//   m2 = relu(m1 @ W2 + b2)          (4 K-chunks, A from LDS scratch)
//   S[node] = sum_k vmask_k * m2_k   (vmask folded; W3 applied in upd)
// Block = 4 waves = 4 nodes; grid = 2000.
// ====================================================================
__global__ __launch_bounds__(128) void msg_kernel(
    const unsigned short* __restrict__ hb,   // [NR][NH] bf16 current h
    const unsigned short* __restrict__ hSb,  // [NR][NH] bf16
    const unsigned short* __restrict__ heb,  // [NE][NH] bf16
    const int* __restrict__ eidx,            // [NR][NK]
    const float* __restrict__ vmask,         // [NR][NK]
    const unsigned short* __restrict__ W1p,  // packed [16][8][32][16]
    const unsigned short* __restrict__ W2p,  // packed [4][8][32][16]
    const float* __restrict__ b1, const float* __restrict__ b2,
    float* __restrict__ S) {                 // [NR][NH]
    __shared__ unsigned short sW[4096];      // 8KB weight chunk (shared by 4 waves)
    __shared__ unsigned short sM[4][NK * NH]; // 4 x 8KB m1 scratch

    const int wave = threadIdx.x >> 5, lane = threadIdx.x & 31;
    const int node = blockIdx.x * 4 + wave;
    const int bIdx = node / NN;
    const int lm = lane & 15, hi = lane >> 4, o1 = hi * 8;

    const int i0 = eidx[node * NK + lm];        // neighbor row for tile0
    const int i1 = eidx[node * NK + 16 + lm];   // neighbor row for tile1
    const size_t nb0 = (size_t)(bIdx * NN + i0) * NH;
    const size_t nb1 = (size_t)(bIdx * NN + i1) * NH;
    const unsigned short* hrow = hb  + (size_t)node * NH;
    const unsigned short* he0  = heb + ((size_t)node * NK + lm) * NH;
    const unsigned short* he1  = heb + ((size_t)node * NK + 16 + lm) * NH;

    v8f acc[2][8];
#pragma unroll
    for (int m = 0; m < 2; ++m)
#pragma unroll
        for (int nc = 0; nc < 8; ++nc) acc[m][nc] = v8f_zero();

    // -------- GEMM1: Kdim = 512, 16 chunks --------
    for (int kc = 0; kc < 16; ++kc) {
        __syncthreads();
        stage8k(sW, W1p + (size_t)kc * 4096, threadIdx.x);   // async DMA to LDS
        // prefetch next chunk into cache while this one is consumed
        if (kc + 1 < 16)
            __builtin_prefetch(W1p + (size_t)(kc + 1) * 4096 + (threadIdx.x << 5), 0, 0);
        __syncthreads();
        const int seg = kc >> 2;
        const int co  = (kc & 3) * 32 + o1;
        const unsigned short *p0, *p1;
        if      (seg == 0) { p0 = hrow;      p1 = hrow;      }  // h broadcast
        else if (seg == 1) { p0 = hb + nb0;  p1 = hb + nb1;  }  // nei_v
        else if (seg == 2) { p0 = hSb + nb0; p1 = hSb + nb1; }  // nei_s
        else               { p0 = he0;       p1 = he1;       }  // h_e
        BF16x16 a0, a1;
        a0.q[0] = ld16(p0 + co); a0.q[1] = ld16(p0 + co + 16);
        a1.q[0] = ld16(p1 + co); a1.q[1] = ld16(p1 + co + 16);
#pragma unroll
        for (int nc = 0; nc < 8; ++nc) {
            BF16x16 bt;
            const unsigned short* wp = sW + (nc * 32 + lane) * 16;
            bt.q[0] = *(const u32x4*)wp; bt.q[1] = *(const u32x4*)(wp + 8);
            acc[0][nc] = wmma_bf16(a0, bt, acc[0][nc]);
            acc[1][nc] = wmma_bf16(a1, bt, acc[1][nc]);
        }
    }

    // epilogue 1: bias + relu -> LDS scratch (row-major bf16 [32][128])
    unsigned short* sm = &sM[wave][0];
    {
        float b1v[8];
#pragma unroll
        for (int nc = 0; nc < 8; ++nc) b1v[nc] = b1[nc * 16 + lm];
#pragma unroll
        for (int m = 0; m < 2; ++m)
#pragma unroll
            for (int v = 0; v < 8; ++v) {
                const int row = m * 16 + hi * 8 + v;
#pragma unroll
                for (int nc = 0; nc < 8; ++nc) {
                    float x = acc[m][nc][v] + b1v[nc];
                    x = fmaxf(x, 0.f);
                    sm[row * NH + nc * 16 + lm] = f2bf(x);
                }
            }
    }

    // -------- GEMM2: Kdim = 128, 4 chunks --------
    v8f acc2[2][8];
#pragma unroll
    for (int m = 0; m < 2; ++m)
#pragma unroll
        for (int nc = 0; nc < 8; ++nc) acc2[m][nc] = v8f_zero();

    for (int kc = 0; kc < 4; ++kc) {
        __syncthreads();   // also protects sW reads of previous chunk
        stage8k(sW, W2p + (size_t)kc * 4096, threadIdx.x);
        if (kc + 1 < 4)
            __builtin_prefetch(W2p + (size_t)(kc + 1) * 4096 + (threadIdx.x << 5), 0, 0);
        __syncthreads();
        const int co = kc * 32 + o1;
        const unsigned short* m0 = sm + lm * NH;
        const unsigned short* m1 = sm + (16 + lm) * NH;
        BF16x16 a0, a1;
        a0.q[0] = *(const u32x4*)(m0 + co); a0.q[1] = *(const u32x4*)(m0 + co + 16);
        a1.q[0] = *(const u32x4*)(m1 + co); a1.q[1] = *(const u32x4*)(m1 + co + 16);
#pragma unroll
        for (int nc = 0; nc < 8; ++nc) {
            BF16x16 bt;
            const unsigned short* wp = sW + (nc * 32 + lane) * 16;
            bt.q[0] = *(const u32x4*)wp; bt.q[1] = *(const u32x4*)(wp + 8);
            acc2[0][nc] = wmma_bf16(a0, bt, acc2[0][nc]);
            acc2[1][nc] = wmma_bf16(a1, bt, acc2[1][nc]);
        }
    }

    // epilogue 2: bias + relu, vmask-weighted reduction over the 32 rows
    {
        float b2v[8];
#pragma unroll
        for (int nc = 0; nc < 8; ++nc) b2v[nc] = b2[nc * 16 + lm];
        float vm[2][8];
#pragma unroll
        for (int m = 0; m < 2; ++m)
#pragma unroll
            for (int v = 0; v < 8; ++v)
                vm[m][v] = vmask[node * NK + m * 16 + hi * 8 + v];
        float p[8];
#pragma unroll
        for (int nc = 0; nc < 8; ++nc) p[nc] = 0.f;
#pragma unroll
        for (int m = 0; m < 2; ++m)
#pragma unroll
            for (int nc = 0; nc < 8; ++nc)
#pragma unroll
                for (int v = 0; v < 8; ++v) {
                    float x = fmaxf(acc2[m][nc][v] + b2v[nc], 0.f);
                    p[nc] += x * vm[m][v];
                }
#pragma unroll
        for (int nc = 0; nc < 8; ++nc) p[nc] += __shfl_xor(p[nc], 16);
        if (hi == 0) {
#pragma unroll
            for (int nc = 0; nc < 8; ++nc)
                S[(size_t)node * NH + nc * 16 + lm] = p[nc];
        }
    }
}

// ====================================================================
// upd: dh = (S @ W3 + sumv*b3)/30 ; h' = LN(h + dh)*mask  (exact fold)
// ====================================================================
__global__ __launch_bounds__(128) void upd_kernel(
    const float* __restrict__ S, const unsigned short* __restrict__ W3p,
    const float* __restrict__ b3, const float* __restrict__ gamma,
    const float* __restrict__ beta,
    const float* __restrict__ sumv, const float* __restrict__ mask,
    const float* __restrict__ h_in, float* __restrict__ h_out,
    unsigned short* __restrict__ hb_out) {
    const int wave = threadIdx.x >> 5, lane = threadIdx.x & 31;
    const int r0 = (blockIdx.x * 4 + wave) * 32;
    if (r0 >= NR) return;
    const int lm = lane & 15, hi = lane >> 4, o1 = hi * 8;

    v8f acc[2][8];
#pragma unroll
    for (int m = 0; m < 2; ++m)
#pragma unroll
        for (int nc = 0; nc < 8; ++nc) acc[m][nc] = v8f_zero();

    const float* xr0 = S + (size_t)(r0 + lm) * NH;
    const float* xr1 = S + (size_t)(r0 + 16 + lm) * NH;
#pragma unroll
    for (int kc = 0; kc < 4; ++kc) {
        const int co = kc * 32 + o1;
        BF16x16 a0, a1;
        a_from_f32(a0, xr0, co);
        a_from_f32(a1, xr1, co);
#pragma unroll
        for (int nc = 0; nc < 8; ++nc) {
            BF16x16 bt;
            const unsigned short* wp = W3p + (size_t)((kc * 8 + nc) * 32 + lane) * 16;
            bt.q[0] = ld16(wp); bt.q[1] = ld16(wp + 8);
            acc[0][nc] = wmma_bf16(a0, bt, acc[0][nc]);
            acc[1][nc] = wmma_bf16(a1, bt, acc[1][nc]);
        }
    }
    float b3v[8], gv[8], bev[8];
#pragma unroll
    for (int nc = 0; nc < 8; ++nc) {
        int c = nc * 16 + lm;
        b3v[nc] = b3[c]; gv[nc] = gamma[c]; bev[nc] = beta[c];
    }
    float sv[2][8], mk[2][8];
#pragma unroll
    for (int m = 0; m < 2; ++m)
#pragma unroll
        for (int v = 0; v < 8; ++v) {
            const int row = r0 + m * 16 + hi * 8 + v;
            sv[m][v] = sumv[row];
            mk[m][v] = mask[row];
        }
    // residual + dh/SCALE
#pragma unroll
    for (int m = 0; m < 2; ++m)
#pragma unroll
        for (int v = 0; v < 8; ++v) {
            const int row = r0 + m * 16 + hi * 8 + v;
#pragma unroll
            for (int nc = 0; nc < 8; ++nc) {
                const int c = nc * 16 + lm;
                acc[m][nc][v] = h_in[(size_t)row * NH + c] +
                                (acc[m][nc][v] + sv[m][v] * b3v[nc]) * SCALE_INV;
            }
        }
    // LayerNorm + mask
#pragma unroll
    for (int m = 0; m < 2; ++m) {
#pragma unroll
        for (int v = 0; v < 8; ++v) {
            float s = 0.f;
#pragma unroll
            for (int nc = 0; nc < 8; ++nc) s += acc[m][nc][v];
            s += __shfl_xor(s, 1); s += __shfl_xor(s, 2);
            s += __shfl_xor(s, 4); s += __shfl_xor(s, 8);
            const float mu = s * (1.0f / 128.0f);
            float q = 0.f;
#pragma unroll
            for (int nc = 0; nc < 8; ++nc) { float d = acc[m][nc][v] - mu; q += d * d; }
            q += __shfl_xor(q, 1); q += __shfl_xor(q, 2);
            q += __shfl_xor(q, 4); q += __shfl_xor(q, 8);
            const float sig  = sqrtf(q * (1.0f / 127.0f) + LN_EPS);
            const float rstd = 1.0f / (sig + LN_EPS);
            const int row = r0 + m * 16 + hi * 8 + v;
#pragma unroll
            for (int nc = 0; nc < 8; ++nc) {
                const int c = nc * 16 + lm;
                float y = gv[nc] * (acc[m][nc][v] - mu) * rstd + bev[nc];
                y *= mk[m][v];
                h_out[(size_t)row * NH + c] = y;
                hb_out[(size_t)row * NH + c] = f2bf(y);
            }
        }
    }
}

// ====================================================================
// launch
// ====================================================================
extern "C" void kernel_launch(void* const* d_in, const int* in_sizes, int n_in,
                              void* d_out, int out_size, void* d_ws, size_t ws_size,
                              hipStream_t stream) {
    const float* V      = (const float*)d_in[0];
    const float* E      = (const float*)d_in[1];
    const float* hS     = (const float*)d_in[2];
    const int*   E_idx  = (const int*)d_in[3];
    const float* mask   = (const float*)d_in[4];
    const float* Wv_w   = (const float*)d_in[5];
    const float* Wv_b   = (const float*)d_in[6];
    const float* Wv_g   = (const float*)d_in[7];
    const float* Wv_be  = (const float*)d_in[8];
    const float* We_w   = (const float*)d_in[9];
    const float* We_b   = (const float*)d_in[10];
    const float* We_g   = (const float*)d_in[11];
    const float* We_be  = (const float*)d_in[12];
    const float* L_W1   = (const float*)d_in[13];
    const float* L_b1   = (const float*)d_in[14];
    const float* L_W2   = (const float*)d_in[15];
    const float* L_b2   = (const float*)d_in[16];
    const float* L_W3   = (const float*)d_in[17];
    const float* L_b3   = (const float*)d_in[18];
    const float* L_g    = (const float*)d_in[19];
    const float* L_be   = (const float*)d_in[20];

    // ---- workspace carve-up (~84 MB total) ----
    char* wsp = (char*)d_ws;
    auto carve = [&](size_t bytes) -> char* {
        char* p = wsp; wsp += (bytes + 255) & ~(size_t)255; return p;
    };
    unsigned short* Wv_p = (unsigned short*)carve(16384 * 2);
    unsigned short* We_p = (unsigned short*)carve(16384 * 2);
    unsigned short* W1_p = (unsigned short*)carve((size_t)3 * 65536 * 2);
    unsigned short* W2_p = (unsigned short*)carve((size_t)3 * 16384 * 2);
    unsigned short* W3_p = (unsigned short*)carve((size_t)3 * 16384 * 2);
    unsigned short* hSb  = (unsigned short*)carve((size_t)NR * NH * 2);
    unsigned short* heb  = (unsigned short*)carve((size_t)NE * NH * 2);
    unsigned short* hb   = (unsigned short*)carve((size_t)NR * NH * 2);
    float* hA    = (float*)carve((size_t)NR * NH * 4);
    float* hBuf  = (float*)carve((size_t)NR * NH * 4);
    float* Sbuf  = (float*)carve((size_t)NR * NH * 4);
    float* vmask = (float*)carve((size_t)NR * NK * 4);
    float* sumv  = (float*)carve((size_t)NR * 4);

    // ---- pre-pack weights (bf16, B-tile layout) ----
    pack_w_kernel<<<(16384 + 255) / 256, 256, 0, stream>>>(Wv_w, Wv_p, 4);
    pack_w_kernel<<<(16384 + 255) / 256, 256, 0, stream>>>(We_w, We_p, 4);
    for (int l = 0; l < 3; ++l) {
        pack_w_kernel<<<(65536 + 255) / 256, 256, 0, stream>>>(
            L_W1 + (size_t)l * 512 * NH, W1_p + (size_t)l * 65536, 16);
        pack_w_kernel<<<(16384 + 255) / 256, 256, 0, stream>>>(
            L_W2 + (size_t)l * NH * NH, W2_p + (size_t)l * 16384, 4);
        pack_w_kernel<<<(16384 + 255) / 256, 256, 0, stream>>>(
            L_W3 + (size_t)l * NH * NH, W3_p + (size_t)l * 16384, 4);
    }
    // ---- bf16 copy of hS, mask gather ----
    cvt_bf16_kernel<<<(NR * NH + 255) / 256, 256, 0, stream>>>(hS, hSb, NR * NH);
    mask_setup_kernel<<<NR, 32, 0, stream>>>(E_idx, mask, vmask, sumv);

    // ---- h_v = LN(V@Wv+b) -> hA(f32) + hb(bf16);  h_e = LN(E@We+b) -> heb ----
    proj_ln_kernel<<<(NR + 127) / 128, 128, 0, stream>>>(
        V, NR, Wv_p, Wv_b, Wv_g, Wv_be, hA, hb);
    proj_ln_kernel<<<NE / 128, 128, 0, stream>>>(
        E, NE, We_p, We_b, We_g, We_be, nullptr, heb);

    // ---- 3 message-passing layers (ping-pong f32 h) ----
    float* hin = hA;
    float* hpong[2] = {hBuf, hA};
    for (int l = 0; l < 3; ++l) {
        msg_kernel<<<NR / 4, 128, 0, stream>>>(
            hb, hSb, heb, E_idx, vmask,
            W1_p + (size_t)l * 65536, W2_p + (size_t)l * 16384,
            L_b1 + l * NH, L_b2 + l * NH, Sbuf);
        float* hout = (l == 2) ? (float*)d_out : hpong[l & 1];
        upd_kernel<<<(NR + 127) / 128, 128, 0, stream>>>(
            Sbuf, W3_p + (size_t)l * 16384,
            L_b3 + l * NH, L_g + l * NH, L_be + l * NH,
            sumv, mask, hin, hout, hb);
        hin = hout;
    }
    (void)in_sizes; (void)n_in; (void)out_size; (void)ws_size;
}
